// R_NEM_6743098655434
// MI455X (gfx1250) — matrix operand order repaired and checked by reading, not compiled, not run on hardware
//
#include <hip/hip_runtime.h>

// ---------------------------------------------------------------------------
// MI455X / gfx1250, wave32. BF16 WMMA (16x16x32, f32 accum) tiled GEMMs with
// fused LayerNorm epilogues. Fragment operands staged in LDS as bf16 with
// bank-conflict-free padded rows; fragments loaded as ds_load_b128 pairs.
// ---------------------------------------------------------------------------

typedef __attribute__((ext_vector_type(16))) __bf16 v16bf;
typedef __attribute__((ext_vector_type(8)))  __bf16 v8bf;
typedef __attribute__((ext_vector_type(2)))  __bf16 v2bf;
typedef __attribute__((ext_vector_type(8)))  float  v8f;

#define KCOUNT   8
#define FC       256
#define LASTN    128
#define BATCH    2048
#define MDIM     1024
#define HDIM     512
#define NROWS    (BATCH * KCOUNT)          /* 16384  */
#define NPROWS   (NROWS * (KCOUNT - 1))    /* 114688 */

// Padded LDS tile row: 40 bf16 = 80 B. 80 % 16 == 0 keeps b128 alignment,
// and the 20-dword bank stride maps lanes 0..15 to 16 distinct banks.
#define APAD 40
#define BPAD 40

// ---- WMMA fragment builders (ISA 7.12.2 layouts, wave32) -------------------
// A (16x32 bf16), Ab row-major [16][APAD]: lane m=lane&15, h=lane>>4 reads
// K runs [h*8, h*8+8) and [16+h*8, 16+h*8+8) -> two aligned b128 loads.
static __device__ __forceinline__ v16bf frag_a(const __bf16* Ab, int lane) {
  const int m = lane & 15, h = lane >> 4;
  const v8bf lo = *(const v8bf*)(Ab + m * APAD + h * 8);
  const v8bf hi = *(const v8bf*)(Ab + m * APAD + 16 + h * 8);
  return __builtin_shufflevector(lo, hi, 0, 1, 2, 3, 4, 5, 6, 7,
                                 8, 9, 10, 11, 12, 13, 14, 15);
}
// B (32x16 bf16), staged TRANSPOSED as Bbt[n][k] ([N][BPAD]): lane n, h=lane>>4
// reads K runs [h*8, h*8+8) and [16+h*8, 16+h*8+8) -> two aligned b128 loads.
static __device__ __forceinline__ v16bf frag_b(const __bf16* Bbt, int nbase,
                                               int lane) {
  const int n  = nbase + (lane & 15);
  const int kb = (lane >> 4) * 8;
  const v8bf lo = *(const v8bf*)(Bbt + n * BPAD + kb);
  const v8bf hi = *(const v8bf*)(Bbt + n * BPAD + kb + 16);
  return __builtin_shufflevector(lo, hi, 0, 1, 2, 3, 4, 5, 6, 7,
                                 8, 9, 10, 11, 12, 13, 14, 15);
}
// C/D (16x16 f32): VGPR v holds row (lane>>4)*8+v, col lane&15.
static __device__ __forceinline__ void store_c(float* Ob, int ldo, int nbase,
                                               int lane, v8f c) {
  const int n = nbase + (lane & 15), mh = lane >> 4;
#pragma unroll
  for (int v = 0; v < 8; ++v) Ob[(mh * 8 + v) * ldo + n] = c[v];
}

// Stage a 32xFC f32 weight tile transposed into Bbt (bf16). Coalesced float4
// reads along n; packed v2bf (K-pair) b32 writes. 4 units/thread.
static __device__ __forceinline__ void stage_w256(const float* __restrict__ w,
                                                  int k0, __bf16* Bbt, int tid) {
#pragma unroll
  for (int t = 0; t < 4; ++t) {
    const int u = tid + t * 256;
    const int k = (u >> 6) * 2;        // 0,2,..,30
    const int n = (u & 63) * 4;        // 0,4,..,252
    const float4 r0 = *(const float4*)(w + (size_t)(k0 + k) * FC + n);
    const float4 r1 = *(const float4*)(w + (size_t)(k0 + k + 1) * FC + n);
    v2bf q0; q0[0] = (__bf16)r0.x; q0[1] = (__bf16)r1.x;
    v2bf q1; q1[0] = (__bf16)r0.y; q1[1] = (__bf16)r1.y;
    v2bf q2; q2[0] = (__bf16)r0.z; q2[1] = (__bf16)r1.z;
    v2bf q3; q3[0] = (__bf16)r0.w; q3[1] = (__bf16)r1.w;
    *(v2bf*)(Bbt + (n + 0) * BPAD + k) = q0;
    *(v2bf*)(Bbt + (n + 1) * BPAD + k) = q1;
    *(v2bf*)(Bbt + (n + 2) * BPAD + k) = q2;
    *(v2bf*)(Bbt + (n + 3) * BPAD + k) = q3;
  }
}
// Same for a 32xLASTN tile (2 units/thread).
static __device__ __forceinline__ void stage_w128(const float* __restrict__ w,
                                                  int k0, __bf16* Bbt, int tid) {
#pragma unroll
  for (int t = 0; t < 2; ++t) {
    const int u = tid + t * 256;
    const int k = (u >> 5) * 2;
    const int n = (u & 31) * 4;
    const float4 r0 = *(const float4*)(w + (size_t)(k0 + k) * LASTN + n);
    const float4 r1 = *(const float4*)(w + (size_t)(k0 + k + 1) * LASTN + n);
    v2bf q0; q0[0] = (__bf16)r0.x; q0[1] = (__bf16)r1.x;
    v2bf q1; q1[0] = (__bf16)r0.y; q1[1] = (__bf16)r1.y;
    v2bf q2; q2[0] = (__bf16)r0.z; q2[1] = (__bf16)r1.z;
    v2bf q3; q3[0] = (__bf16)r0.w; q3[1] = (__bf16)r1.w;
    *(v2bf*)(Bbt + (n + 0) * BPAD + k) = q0;
    *(v2bf*)(Bbt + (n + 1) * BPAD + k) = q1;
    *(v2bf*)(Bbt + (n + 2) * BPAD + k) = q2;
    *(v2bf*)(Bbt + (n + 3) * BPAD + k) = q3;
  }
}

// Row mean / rstd via 16-lane butterfly (xor 1,2,4,8 stays in half-wave).
static __device__ __forceinline__ float2 rowstats(const float* Ob, int ldo,
                                                  int r, int c0, int ntot) {
  float s = 0.f, ss = 0.f;
  for (int c = c0; c < ntot; c += 16) {
    float v = Ob[r * ldo + c];
    s += v; ss += v * v;
  }
#pragma unroll
  for (int off = 1; off < 16; off <<= 1) {
    s  += __shfl_xor(s,  off, 32);
    ss += __shfl_xor(ss, off, 32);
  }
  const float inv = 1.f / (float)ntot;
  const float mu  = s * inv;
  const float var = ss * inv - mu * mu;
  float2 out; out.x = mu; out.y = rsqrtf(var + 1e-5f);
  return out;
}

// ---------------------------------------------------------------------------
// K1: s1 = relu(LN(state @ enc_w + enc_b))   [16384x512]x[512x256] -> bf16
// ---------------------------------------------------------------------------
__global__ __launch_bounds__(256) void enc_kernel(
    const float* __restrict__ state, const float* __restrict__ w,
    const float* __restrict__ bias, const float* __restrict__ g,
    const float* __restrict__ bb, __bf16* __restrict__ s1) {
  __shared__ __align__(16) __bf16 Ab[16 * APAD];
  __shared__ __align__(16) __bf16 Bbt[FC * BPAD];
  __shared__ float Ob[16 * FC];
  const int tid = threadIdx.x, lane = tid & 31, wave = tid >> 5;
  const int row0 = blockIdx.x * 16;
  const int nb0 = wave * 32, nb1 = nb0 + 16;
  v8f acc0 = {}; v8f acc1 = {};

  for (int k0 = 0; k0 < HDIM; k0 += 32) {
    {   // stage A: f32 -> bf16 pair (one b32 LDS store)
      const int rA = tid >> 4, kk = (tid & 15) * 2;
      const float2 v = *(const float2*)(state + (size_t)(row0 + rA) * HDIM + k0 + kk);
      v2bf p; p[0] = (__bf16)v.x; p[1] = (__bf16)v.y;
      *(v2bf*)(Ab + rA * APAD + kk) = p;
    }
    stage_w256(w, k0, Bbt, tid);
    __syncthreads();
    v16bf a  = frag_a(Ab, lane);
    v16bf b0 = frag_b(Bbt, nb0, lane);
    v16bf b1 = frag_b(Bbt, nb1, lane);
    acc0 = __builtin_amdgcn_wmma_f32_16x16x32_bf16(false, a, false, b0, (short)0, acc0, false, false);
    acc1 = __builtin_amdgcn_wmma_f32_16x16x32_bf16(false, a, false, b1, (short)0, acc1, false, false);
    __syncthreads();
    if (k0 + 32 < HDIM)
      __builtin_prefetch(w + (size_t)(k0 + 32) * FC + (tid & 255), 0, 1);
  }
  store_c(Ob, FC, nb0, lane, acc0);
  store_c(Ob, FC, nb1, lane, acc1);
  __syncthreads();
  const int r = tid >> 4, c0 = tid & 15;
  for (int c = c0; c < FC; c += 16) Ob[r * FC + c] += bias[c];
  const float2 st = rowstats(Ob, FC, r, c0, FC);
  for (int c = c0; c < FC; c += 16) {
    float v = (Ob[r * FC + c] - st.x) * st.y * g[c] + bb[c];
    s1[(size_t)(row0 + r) * FC + c] = (__bf16)fmaxf(v, 0.f);
  }
}

// ---------------------------------------------------------------------------
// K2: core = relu(LN(concat(s1_self, s1_other) @ core_w + b))
//     A gathered on the fly from bf16 s1 (no 235MB concat materialization).
// ---------------------------------------------------------------------------
__global__ __launch_bounds__(256) void core_kernel(
    const __bf16* __restrict__ s1, const float* __restrict__ w,
    const float* __restrict__ bias, const float* __restrict__ g,
    const float* __restrict__ bb, __bf16* __restrict__ core) {
  __shared__ __align__(16) __bf16 Ab[16 * APAD];
  __shared__ __align__(16) __bf16 Bbt[FC * BPAD];
  __shared__ float Ob[16 * FC];
  const int tid = threadIdx.x, lane = tid & 31, wave = tid >> 5;
  const int row0 = blockIdx.x * 16;
  const int nb0 = wave * 32, nb1 = nb0 + 16;

  // pair decomposition for this staging thread's row
  const int rA = tid >> 4;
  const int p  = row0 + rA;
  const int q  = p / 7, jj = p % 7;
  const int bI = q >> 3, i = q & 7;
  const int j  = jj + (jj >= i ? 1 : 0);
  const int rowSelf = q, rowOther = (bI << 3) + j;

  v8f acc0 = {}; v8f acc1 = {};
  for (int k0 = 0; k0 < 2 * FC; k0 += 32) {   // K-tiles never straddle FC
    {
      const int kk = (tid & 15) * 2;
      const __bf16* src = (k0 < FC) ? (s1 + (size_t)rowSelf * FC + k0)
                                    : (s1 + (size_t)rowOther * FC + (k0 - FC));
      *(v2bf*)(Ab + rA * APAD + kk) = *(const v2bf*)(src + kk);
    }
    stage_w256(w, k0, Bbt, tid);
    __syncthreads();
    v16bf a  = frag_a(Ab, lane);
    v16bf b0 = frag_b(Bbt, nb0, lane);
    v16bf b1 = frag_b(Bbt, nb1, lane);
    acc0 = __builtin_amdgcn_wmma_f32_16x16x32_bf16(false, a, false, b0, (short)0, acc0, false, false);
    acc1 = __builtin_amdgcn_wmma_f32_16x16x32_bf16(false, a, false, b1, (short)0, acc1, false, false);
    __syncthreads();
    if (k0 + 32 < 2 * FC)
      __builtin_prefetch(w + (size_t)(k0 + 32) * FC + (tid & 255), 0, 1);
  }
  store_c(Ob, FC, nb0, lane, acc0);
  store_c(Ob, FC, nb1, lane, acc1);
  __syncthreads();
  const int r = tid >> 4, c0 = tid & 15;
  for (int c = c0; c < FC; c += 16) Ob[r * FC + c] += bias[c];
  const float2 st = rowstats(Ob, FC, r, c0, FC);
  for (int c = c0; c < FC; c += 16) {
    float v = (Ob[r * FC + c] - st.x) * st.y * g[c] + bb[c];
    core[(size_t)(row0 + r) * FC + c] = (__bf16)fmaxf(v, 0.f);
  }
}

// ---------------------------------------------------------------------------
// K3: att = sigmoid(tanh(LN(core @ att1_w + b)) @ att2_w + b2)  (N=128)
// ---------------------------------------------------------------------------
__global__ __launch_bounds__(256) void att_kernel(
    const __bf16* __restrict__ core, const float* __restrict__ w1,
    const float* __restrict__ b1, const float* __restrict__ g,
    const float* __restrict__ bb, const float* __restrict__ w2,
    const float* __restrict__ b2, float* __restrict__ att) {
  __shared__ __align__(16) __bf16 Ab[16 * APAD];
  __shared__ __align__(16) __bf16 Bbt[LASTN * BPAD];
  __shared__ float Ob[16 * LASTN];
  const int tid = threadIdx.x, lane = tid & 31, wave = tid >> 5;
  const int row0 = blockIdx.x * 16;
  const int nb = wave * 16;                      // 8 waves x 1 tile = 128 cols
  v8f acc = {};

  for (int k0 = 0; k0 < FC; k0 += 32) {
    {
      const int rA = tid >> 4, kk = (tid & 15) * 2;
      const __bf16* src = core + (size_t)(row0 + rA) * FC + k0;
      *(v2bf*)(Ab + rA * APAD + kk) = *(const v2bf*)(src + kk);
    }
    stage_w128(w1, k0, Bbt, tid);
    __syncthreads();
    v16bf a = frag_a(Ab, lane);
    v16bf b = frag_b(Bbt, nb, lane);
    acc = __builtin_amdgcn_wmma_f32_16x16x32_bf16(false, a, false, b, (short)0, acc, false, false);
    __syncthreads();
  }
  store_c(Ob, LASTN, nb, lane, acc);
  __syncthreads();
  const int r = tid >> 4, c0 = tid & 15;
  for (int c = c0; c < LASTN; c += 16) Ob[r * LASTN + c] += b1[c];
  const float2 st = rowstats(Ob, LASTN, r, c0, LASTN);
  float dot = 0.f;
  for (int c = c0; c < LASTN; c += 16) {
    float v = tanhf((Ob[r * LASTN + c] - st.x) * st.y * g[c] + bb[c]);
    dot += v * w2[c];
  }
#pragma unroll
  for (int off = 1; off < 16; off <<= 1) dot += __shfl_xor(dot, off, 32);
  if (c0 == 0) {
    const float z = dot + b2[0];
    att[row0 + r] = 1.f / (1.f + __expf(-z));
  }
}

// ---------------------------------------------------------------------------
// K4: ctx = relu(LN(core @ ctx_w + b)); effect[g] = sum_7 ctx*att  (fused)
//     One WG handles 112 pair-rows = 16 effect groups; ctx never hits HBM.
// ---------------------------------------------------------------------------
__global__ __launch_bounds__(256) void ctx_effect_kernel(
    const __bf16* __restrict__ core, const float* __restrict__ w,
    const float* __restrict__ bias, const float* __restrict__ g,
    const float* __restrict__ bb, const float* __restrict__ att,
    __bf16* __restrict__ effect) {
  __shared__ __align__(16) __bf16 Ab[16 * APAD];
  __shared__ __align__(16) __bf16 Bbt[FC * BPAD];
  __shared__ float Ob[16 * FC];
  __shared__ float Eff[16 * FC];
  const int tid = threadIdx.x, lane = tid & 31, wave = tid >> 5;
  const int nb0 = wave * 32, nb1 = nb0 + 16;
  const int prow0 = blockIdx.x * 112;
  const int grow0 = blockIdx.x * 16;

  for (int c = tid; c < 16 * FC; c += 256) Eff[c] = 0.f;
  __syncthreads();

  for (int mt = 0; mt < 7; ++mt) {
    const int row0 = prow0 + mt * 16;
    v8f acc0 = {}; v8f acc1 = {};
    for (int k0 = 0; k0 < FC; k0 += 32) {
      __syncthreads();                 // protect LDS from prior-phase readers
      {
        const int rA = tid >> 4, kk = (tid & 15) * 2;
        const __bf16* src = core + (size_t)(row0 + rA) * FC + k0;
        *(v2bf*)(Ab + rA * APAD + kk) = *(const v2bf*)(src + kk);
      }
      stage_w256(w, k0, Bbt, tid);
      __syncthreads();
      v16bf a  = frag_a(Ab, lane);
      v16bf b0 = frag_b(Bbt, nb0, lane);
      v16bf b1 = frag_b(Bbt, nb1, lane);
      acc0 = __builtin_amdgcn_wmma_f32_16x16x32_bf16(false, a, false, b0, (short)0, acc0, false, false);
      acc1 = __builtin_amdgcn_wmma_f32_16x16x32_bf16(false, a, false, b1, (short)0, acc1, false, false);
    }
    __syncthreads();
    store_c(Ob, FC, nb0, lane, acc0);
    store_c(Ob, FC, nb1, lane, acc1);
    __syncthreads();
    const int r = tid >> 4, c0 = tid & 15;
    for (int c = c0; c < FC; c += 16) Ob[r * FC + c] += bias[c];
    const float2 st = rowstats(Ob, FC, r, c0, FC);
    const float a_p = att[row0 + r];
    const int gl = (mt * 16 + r) / 7;          // local effect group 0..15
    for (int c = c0; c < FC; c += 16) {
      float v = fmaxf((Ob[r * FC + c] - st.x) * st.y * g[c] + bb[c], 0.f);
      atomicAdd(&Eff[gl * FC + c], v * a_p);   // ds_add_f32
    }
  }
  __syncthreads();
  {
    const int r = tid >> 4, c0 = tid & 15;
    for (int c = c0; c < FC; c += 16)
      effect[(size_t)(grow0 + r) * FC + c] = (__bf16)Eff[r * FC + c];
  }
}

// ---------------------------------------------------------------------------
// K5: new_state = [s1 | effect | x] @ out_w + out_b   (K=1536, gathered A)
//     Output duplicated (reference returns the tuple (new_state, new_state)).
// ---------------------------------------------------------------------------
__global__ __launch_bounds__(256) void out_kernel(
    const __bf16* __restrict__ s1, const __bf16* __restrict__ effect,
    const float* __restrict__ x, const float* __restrict__ w,
    const float* __restrict__ bias, float* __restrict__ out) {
  __shared__ __align__(16) __bf16 Ab[16 * APAD];
  __shared__ __align__(16) __bf16 Bbt[FC * BPAD];
  __shared__ float Ob[16 * FC];
  const int tid = threadIdx.x, lane = tid & 31, wave = tid >> 5;
  const int row0 = blockIdx.x * 16;
  const int nb0 = wave * 32, nb1 = nb0 + 16;
  const int KTOT = FC + FC + MDIM;   // 1536
  v8f acc0 = {}; v8f acc1 = {};

  for (int k0 = 0; k0 < KTOT; k0 += 32) {
    {
      const int rA = tid >> 4, kk = (tid & 15) * 2;
      const int row = row0 + rA;
      if (k0 < FC) {
        *(v2bf*)(Ab + rA * APAD + kk) =
            *(const v2bf*)(s1 + (size_t)row * FC + k0 + kk);
      } else if (k0 < 2 * FC) {
        *(v2bf*)(Ab + rA * APAD + kk) =
            *(const v2bf*)(effect + (size_t)row * FC + (k0 - FC) + kk);
      } else {
        const float2 v = *(const float2*)(x + (size_t)row * MDIM + (k0 - 2 * FC) + kk);
        v2bf pkd; pkd[0] = (__bf16)v.x; pkd[1] = (__bf16)v.y;
        *(v2bf*)(Ab + rA * APAD + kk) = pkd;
      }
    }
    stage_w256(w, k0, Bbt, tid);
    __syncthreads();
    v16bf a  = frag_a(Ab, lane);
    v16bf b0 = frag_b(Bbt, nb0, lane);
    v16bf b1 = frag_b(Bbt, nb1, lane);
    acc0 = __builtin_amdgcn_wmma_f32_16x16x32_bf16(false, a, false, b0, (short)0, acc0, false, false);
    acc1 = __builtin_amdgcn_wmma_f32_16x16x32_bf16(false, a, false, b1, (short)0, acc1, false, false);
    __syncthreads();
    if (k0 + 32 < KTOT)
      __builtin_prefetch(w + (size_t)(k0 + 32) * FC + (tid & 255), 0, 1);
  }
  store_c(Ob, FC, nb0, lane, acc0);
  store_c(Ob, FC, nb1, lane, acc1);
  __syncthreads();
  const int r = tid >> 4, c0 = tid & 15;
  for (int c = c0; c < FC; c += 16) {
    const float v = Ob[r * FC + c] + bias[c];
    out[(size_t)(row0 + r) * FC + c] = v;
    out[(size_t)NROWS * FC + (size_t)(row0 + r) * FC + c] = v;   // tuple copy
  }
}

// ---------------------------------------------------------------------------
extern "C" void kernel_launch(void* const* d_in, const int* in_sizes, int n_in,
                              void* d_out, int out_size, void* d_ws, size_t ws_size,
                              hipStream_t stream) {
  (void)in_sizes; (void)n_in; (void)out_size; (void)ws_size;
  const float* x       = (const float*)d_in[0];
  const float* state   = (const float*)d_in[1];
  const float* enc_w   = (const float*)d_in[2];
  const float* enc_b   = (const float*)d_in[3];
  const float* enc_g   = (const float*)d_in[4];
  const float* enc_bb  = (const float*)d_in[5];
  const float* core_w  = (const float*)d_in[6];
  const float* core_b  = (const float*)d_in[7];
  const float* core_g  = (const float*)d_in[8];
  const float* core_bb = (const float*)d_in[9];
  const float* ctx_w   = (const float*)d_in[10];
  const float* ctx_b   = (const float*)d_in[11];
  const float* ctx_g   = (const float*)d_in[12];
  const float* ctx_bb  = (const float*)d_in[13];
  const float* att1_w  = (const float*)d_in[14];
  const float* att1_b  = (const float*)d_in[15];
  const float* att_g   = (const float*)d_in[16];
  const float* att_bb  = (const float*)d_in[17];
  const float* att2_w  = (const float*)d_in[18];
  const float* att2_b  = (const float*)d_in[19];
  const float* out_w   = (const float*)d_in[20];
  const float* out_b   = (const float*)d_in[21];
  float* out = (float*)d_out;

  char* ws = (char*)d_ws;
  size_t off = 0;
  __bf16* s1   = (__bf16*)(ws + off); off += (size_t)NROWS  * FC * sizeof(__bf16);
  __bf16* core = (__bf16*)(ws + off); off += (size_t)NPROWS * FC * sizeof(__bf16);
  float*  att  = (float*) (ws + off); off += (size_t)NPROWS * sizeof(float);
  __bf16* eff  = (__bf16*)(ws + off); off += (size_t)NROWS  * FC * sizeof(__bf16);

  enc_kernel<<<NROWS / 16, 256, 0, stream>>>(state, enc_w, enc_b, enc_g, enc_bb, s1);
  core_kernel<<<NPROWS / 16, 256, 0, stream>>>(s1, core_w, core_b, core_g, core_bb, core);
  att_kernel<<<NPROWS / 16, 256, 0, stream>>>(core, att1_w, att1_b, att_g, att_bb,
                                              att2_w, att2_b, att);
  ctx_effect_kernel<<<NPROWS / 112, 256, 0, stream>>>(core, ctx_w, ctx_b, ctx_g, ctx_bb,
                                                      att, eff);
  out_kernel<<<NROWS / 16, 256, 0, stream>>>(s1, eff, x, out_w, out_b, out);
}
